// MessagePassingGATLayer_26895085208207
// MI455X (gfx1250) — matrix operand order repaired and checked by reading, not compiled, not run on hardware
//
#include <hip/hip_runtime.h>
#include <hip/hip_bf16.h>
#include <cstdint>

#define ALPHA 0.2f

typedef __attribute__((ext_vector_type(16))) __bf16 v16bf;
typedef __attribute__((ext_vector_type(8)))  float  v8f;

constexpr int Bb = 64;      // batch (== N, reference reshape bug requires it)
constexpr int Nn = 64;      // regulators
constexpr int Mm = 2048;    // targets
constexpr int Kk = 256;     // feature dim
constexpr int Hh = 8;       // heads
constexpr int Ff = 64;      // per-head dim
constexpr int HF = 512;     // H*F

// 32-byte v16bf load from LDS via two 16B chunks (only needs 16B alignment)
__device__ __forceinline__ v16bf lds_ld16bf(const __bf16* p) {
  union { v16bf v; uint4 q[2]; } u;
  u.q[0] = *(const uint4*)(p);
  u.q[1] = *(const uint4*)(p + 8);
  return u.v;
}

// ---------------------------------------------------------------------------
// 1) Fold attention vectors into the projection weights:
//    wrA[k][h] = sum_f W_reg[k, h*F+f] * att[h, f]
//    wtA[k][h] = sum_f W_tgt[k, h*F+f] * att[h, F+f]
// This removes the 34 GFLOP h_tgt GEMM entirely (h_tgt only feeds e_tgt).
// ---------------------------------------------------------------------------
__global__ void k_fold_weights(const float* __restrict__ Wreg,
                               const float* __restrict__ Wtgt,
                               const float* __restrict__ att,
                               float* __restrict__ wrA,
                               float* __restrict__ wtA) {
  int k = threadIdx.x;
  if (k >= Kk) return;
  for (int hh = 0; hh < Hh; ++hh) {
    float s1 = 0.f, s2 = 0.f;
    for (int f = 0; f < Ff; ++f) {
      s1 += Wreg[k * HF + hh * Ff + f] * att[hh * 2 * Ff + f];
      s2 += Wtgt[k * HF + hh * Ff + f] * att[hh * 2 * Ff + Ff + f];
    }
    wrA[k * Hh + hh] = s1;
    wtA[k * Hh + hh] = s2;
  }
}

// ---------------------------------------------------------------------------
// 2) Attention scalars: one wave per feature row (reg rows then tgt rows).
//    er [n][h][b]  (b contiguous -> matches WMMA A-tile k-runs later)
//    etT[h][m][b]
// ---------------------------------------------------------------------------
__global__ void __launch_bounds__(256) k_attn_proj(
    const float* __restrict__ reg_feat, const float* __restrict__ tgt_feat,
    const float* __restrict__ wrA, const float* __restrict__ wtA,
    float* __restrict__ er, float* __restrict__ etT) {
  __shared__ float wl[2][Kk * 9];                 // pad 8->9: conflict-free reads
  for (int i = threadIdx.x; i < Kk * Hh; i += 256) {
    int k = i >> 3, hh = i & 7;
    wl[0][k * 9 + hh] = wrA[i];
    wl[1][k * 9 + hh] = wtA[i];
  }
  __syncthreads();
  const int wave = threadIdx.x >> 5, lane = threadIdx.x & 31;
  const int row = blockIdx.x * 8 + wave;          // 0 .. B*N + B*M - 1
  const bool isReg = row < Bb * Nn;
  const float* feat = isReg ? (reg_feat + (size_t)row * Kk)
                            : (tgt_feat + (size_t)(row - Bb * Nn) * Kk);
  const float* w = isReg ? wl[0] : wl[1];
  float p[Hh] = {0.f, 0.f, 0.f, 0.f, 0.f, 0.f, 0.f, 0.f};
#pragma unroll
  for (int j = 0; j < 8; ++j) {
    int k = lane + 32 * j;                        // coalesced global reads
    float fv = feat[k];
    const float* wr = w + k * 9;
#pragma unroll
    for (int hh = 0; hh < Hh; ++hh) p[hh] += fv * wr[hh];
  }
#pragma unroll
  for (int off = 16; off >= 1; off >>= 1) {
#pragma unroll
    for (int hh = 0; hh < Hh; ++hh) p[hh] += __shfl_xor(p[hh], off, 32);
  }
  if (lane == 0) {
    if (isReg) {
      int b = row >> 6, n = row & 63;
#pragma unroll
      for (int hh = 0; hh < Hh; ++hh) er[(n * Hh + hh) * Bb + b] = p[hh];
    } else {
      int r2 = row - Bb * Nn;
      int b = r2 >> 11, m = r2 & (Mm - 1);
#pragma unroll
      for (int hh = 0; hh < Hh; ++hh)
        etT[((size_t)hh * Mm + m) * Bb + b] = p[hh];
    }
  }
}

// ---------------------------------------------------------------------------
// 3) h_reg = reg_feat @ W_reg as bf16 WMMA GEMM [4096,256]x[256,512].
//    Output layout hrT[n][h][b][f] (bf16) so the main kernel streams a
//    contiguous 8KB [b][f] slab per (n,h).
// ---------------------------------------------------------------------------
__global__ void __launch_bounds__(256) k_hreg_gemm(
    const float* __restrict__ reg_feat, const float* __restrict__ Wreg,
    __bf16* __restrict__ hrT) {
  constexpr int WTP = 264;                        // padded k-stride (bank spread)
  __shared__ __bf16 WT[Ff * WTP];                 // W tile [f_local][k], ~33KB
  const int h = blockIdx.y;                       // 64-wide f-range == head h
  const int tid = threadIdx.x;
#pragma unroll 4
  for (int i = 0; i < 64; ++i) {
    int fl = tid & 63;
    int k = (tid >> 6) + i * 4;
    WT[fl * WTP + k] = (__bf16)Wreg[k * HF + h * Ff + fl]; // coalesced read
  }
  __syncthreads();
  const int wave = tid >> 5, lane = tid & 31;
  const int hi16 = (lane >> 4) & 1, lm = lane & 15;
  const int r = blockIdx.x * 128 + wave * 16 + lm;        // A row (b*64+n)
  v8f zero = {0.f, 0.f, 0.f, 0.f, 0.f, 0.f, 0.f, 0.f};
  v8f acc[4] = {zero, zero, zero, zero};
  for (int ks = 0; ks < 8; ++ks) {
    // A-tile 16x32 bf16: lane layout K = 32*ks + 16*hi16 + 8*run + j
    v16bf av;
#pragma unroll
    for (int run = 0; run < 2; ++run) {
      int k0 = ks * 32 + hi16 * 16 + run * 8;
      float4 p0 = *(const float4*)(reg_feat + (size_t)r * Kk + k0);
      float4 p1 = *(const float4*)(reg_feat + (size_t)r * Kk + k0 + 4);
      av[run * 8 + 0] = (__bf16)p0.x; av[run * 8 + 1] = (__bf16)p0.y;
      av[run * 8 + 2] = (__bf16)p0.z; av[run * 8 + 3] = (__bf16)p0.w;
      av[run * 8 + 4] = (__bf16)p1.x; av[run * 8 + 5] = (__bf16)p1.y;
      av[run * 8 + 6] = (__bf16)p1.z; av[run * 8 + 7] = (__bf16)p1.w;
    }
    int kb = ks * 32 + hi16 * 16;                 // B-tile 32x16: K = 16*hi16 + e
#pragma unroll
    for (int t = 0; t < 4; ++t) {
      v16bf bv = lds_ld16bf(&WT[(t * 16 + lm) * WTP + kb]);
      acc[t] = __builtin_amdgcn_wmma_f32_16x16x32_bf16(
          false, av, false, bv, (short)0, acc[t], false, false);
    }
  }
  const int mb = blockIdx.x * 128 + wave * 16;
#pragma unroll
  for (int t = 0; t < 4; ++t) {
#pragma unroll
    for (int v = 0; v < 8; ++v) {                 // D: row = v + 8*hi16, col = lm
      int row = mb + v + hi16 * 8;
      int bb = row >> 6, nn = row & 63;
      hrT[((size_t)(nn * Hh + h) * Bb + bb) * Ff + t * 16 + lm] =
          (__bf16)acc[t][v];
    }
  }
}

// ---------------------------------------------------------------------------
// 4) Single pass over adj (33.5MB, read once): per (b,m) pack a 64-bit mask
//    over n and compute lse[b,m,h] = rowmax + log(sum exp) of the masked
//    leaky-relu scores (softmax-over-n collapsed to one scalar).
// ---------------------------------------------------------------------------
__global__ void __launch_bounds__(256) k_softmax_stats(
    const int* __restrict__ adj, const float* __restrict__ er,
    const float* __restrict__ etT, float* __restrict__ lseT,
    unsigned long long* __restrict__ adjn) {
  __shared__ float erb[Nn * Hh];                  // er slice for this b
  const int b = blockIdx.y;
  const int m = blockIdx.x * 256 + threadIdx.x;
  for (int i = threadIdx.x; i < Nn * Hh; i += 256) erb[i] = er[i * Bb + b];
  __syncthreads();
  float et[Hh];
#pragma unroll
  for (int hh = 0; hh < Hh; ++hh) et[hh] = etT[((size_t)hh * Mm + m) * Bb + b];
  unsigned long long mask = 0ull;
  float mx[Hh];
#pragma unroll
  for (int hh = 0; hh < Hh; ++hh) mx[hh] = -1e30f;
  const int* arow = adj + (size_t)b * Nn * Mm + m;
  for (int n = 0; n < Nn; ++n) {
    if (arow[(size_t)n * Mm] != 0) {              // coalesced across m
      mask |= (1ull << n);
#pragma unroll
      for (int hh = 0; hh < Hh; ++hh) {
        float e = erb[n * Hh + hh] + et[hh];
        e = (e > 0.f) ? e : ALPHA * e;
        mx[hh] = fmaxf(mx[hh], e);
      }
    }
  }
  float sm[Hh] = {0.f, 0.f, 0.f, 0.f, 0.f, 0.f, 0.f, 0.f};
  for (int n = 0; n < Nn; ++n) {
    if ((mask >> n) & 1ull) {
#pragma unroll
      for (int hh = 0; hh < Hh; ++hh) {
        float e = erb[n * Hh + hh] + et[hh];
        e = (e > 0.f) ? e : ALPHA * e;
        sm[hh] += __expf(e - mx[hh]);
      }
    }
  }
#pragma unroll
  for (int hh = 0; hh < Hh; ++hh)
    lseT[((size_t)hh * Mm + m) * Bb + b] = mx[hh] + __logf(sm[hh]);
  adjn[(size_t)b * Mm + m] = mask;
}

// 5) Bit-transpose adjacency masks: bits-over-n -> bits-over-b (1MB -> 1MB).
__global__ void k_bit_transpose(const unsigned long long* __restrict__ adjn,
                                unsigned long long* __restrict__ adjB) {
  const int n = blockIdx.y;
  const int m = blockIdx.x * 256 + threadIdx.x;
  unsigned long long mask = 0ull;
  for (int b = 0; b < Bb; ++b)
    mask |= ((adjn[(size_t)b * Mm + m] >> n) & 1ull) << b;
  adjB[(size_t)n * Mm + m] = mask;
}

// ---------------------------------------------------------------------------
// 6) Fused attention + message aggregation. Block = (n, h, 128-m tile),
//    wave = one 16-m tile. Per (n,h): out[m,f] = sum_b a[b]*h_reg[b,f] as
//    two k=32 WMMA steps; a recomputed in-register -> bf16 A-tile.
// ---------------------------------------------------------------------------
__global__ void __launch_bounds__(256) k_gat_msgs(
    const __bf16* __restrict__ hrT, const float* __restrict__ er,
    const float* __restrict__ etT, const float* __restrict__ lseT,
    const unsigned long long* __restrict__ adjB, float* __restrict__ out) {
  constexpr int ETP = 68;                         // padded b-stride (floats)
  constexpr int HTP = 72;                         // padded b-stride (bf16)
  __shared__ __bf16 hT[Ff * HTP];                 // h_reg[f][b] for this (n,h)
  __shared__ float er_s[Bb];
  __shared__ float etL[128 * ETP];
  __shared__ float lsL[128 * ETP];

  const int n = blockIdx.z, h = blockIdx.y;
  const int mblk = blockIdx.x * 128;
  const int tid = threadIdx.x;

  {                                               // stage hT transposed
    const __bf16* src = hrT + (size_t)(n * Hh + h) * (Bb * Ff);
#pragma unroll
    for (int j = 0; j < 16; ++j) {
      int s = j * 256 + tid;                      // coalesced over tid
      int b = s >> 6, f = s & 63;
      hT[f * HTP + b] = src[s];
    }
  }
  {                                               // stage e_tgt / lse rows
    const float* etg = etT + ((size_t)h * Mm + mblk) * Bb;
    const float* lsg = lseT + ((size_t)h * Mm + mblk) * Bb;
#pragma unroll
    for (int j = 0; j < 8; ++j) {
      int s = j * 256 + tid;                      // 2048 float4 chunks
      int ml = s >> 4, c = (s & 15) * 4;
      *(float4*)(&etL[ml * ETP + c]) = *(const float4*)(etg + ml * Bb + c);
      *(float4*)(&lsL[ml * ETP + c]) = *(const float4*)(lsg + ml * Bb + c);
    }
  }
  if (tid < Bb) er_s[tid] = er[(n * Hh + h) * Bb + tid];
  __syncthreads();

  const int wave = tid >> 5, lane = tid & 31;
  const int hi16 = (lane >> 4) & 1, lm = lane & 15;
  const int ml = wave * 16 + lm;
  const int m = mblk + ml;
  const unsigned long long amask = adjB[(size_t)n * Mm + m];

  v8f zero = {0.f, 0.f, 0.f, 0.f, 0.f, 0.f, 0.f, 0.f};
  v8f acc[4] = {zero, zero, zero, zero};

#pragma unroll
  for (int ks = 0; ks < 2; ++ks) {
    v16bf av;                                     // A-tile: a[m=lm][k=b]
#pragma unroll
    for (int run = 0; run < 2; ++run) {
      const int b0 = ks * 32 + hi16 * 16 + run * 8;
      float4 e0 = *(const float4*)(&etL[ml * ETP + b0]);
      float4 e1 = *(const float4*)(&etL[ml * ETP + b0 + 4]);
      float4 l0 = *(const float4*)(&lsL[ml * ETP + b0]);
      float4 l1 = *(const float4*)(&lsL[ml * ETP + b0 + 4]);
      float4 r0 = *(const float4*)(&er_s[b0]);
      float4 r1 = *(const float4*)(&er_s[b0 + 4]);
      float ee[8] = {r0.x + e0.x, r0.y + e0.y, r0.z + e0.z, r0.w + e0.w,
                     r1.x + e1.x, r1.y + e1.y, r1.z + e1.z, r1.w + e1.w};
      float ll[8] = {l0.x, l0.y, l0.z, l0.w, l1.x, l1.y, l1.z, l1.w};
      const unsigned bits = (unsigned)((amask >> b0) & 0xffull);
#pragma unroll
      for (int j = 0; j < 8; ++j) {
        float e = ee[j];
        e = (e > 0.f) ? e : ALPHA * e;            // leaky relu
        float a = ((bits >> j) & 1u) ? __expf(e - ll[j]) : 0.f;
        av[run * 8 + j] = (__bf16)a;              // softmax prob (exp(e-lse))
      }
    }
#pragma unroll
    for (int t = 0; t < 4; ++t) {
      v16bf bv = lds_ld16bf(&hT[(t * 16 + lm) * HTP + ks * 32 + hi16 * 16]);
      acc[t] = __builtin_amdgcn_wmma_f32_16x16x32_bf16(
          false, av, false, bv, (short)0, acc[t], false, false);
    }
  }
  const size_t obase = ((size_t)n * Mm) * HF + (size_t)h * Ff;
#pragma unroll
  for (int t = 0; t < 4; ++t) {
#pragma unroll
    for (int v = 0; v < 8; ++v) {
      int mrow = mblk + wave * 16 + v + hi16 * 8;
      out[obase + (size_t)mrow * HF + t * 16 + lm] = acc[t][v];
    }
  }
}

// ---------------------------------------------------------------------------
extern "C" void kernel_launch(void* const* d_in, const int* in_sizes, int n_in,
                              void* d_out, int out_size, void* d_ws,
                              size_t ws_size, hipStream_t stream) {
  (void)in_sizes; (void)n_in; (void)out_size; (void)ws_size;
  const float* reg_feat = (const float*)d_in[0];
  const float* tgt_feat = (const float*)d_in[1];
  const int* adj = (const int*)d_in[2];
  const float* W_reg = (const float*)d_in[3];
  const float* W_tgt = (const float*)d_in[4];
  const float* att = (const float*)d_in[5];
  float* out = (float*)d_out;

  // Workspace carve-up (~14.2 MB, all L2-resident)
  float* wrA = (float*)d_ws;                      // [256][8]
  float* wtA = wrA + Kk * Hh;                     // [256][8]
  float* er = wtA + Kk * Hh;                      // [N][H][B]
  float* etT = er + Nn * Hh * Bb;                 // [H][M][B]
  float* lseT = etT + (size_t)Hh * Mm * Bb;       // [H][M][B]
  unsigned long long* adjn =
      (unsigned long long*)(lseT + (size_t)Hh * Mm * Bb);   // [B][M] bits-over-n
  unsigned long long* adjB = adjn + (size_t)Bb * Mm;        // [N][M] bits-over-b
  __bf16* hrT = (__bf16*)(adjB + (size_t)Nn * Mm);          // [N][H][B][F] bf16

  k_fold_weights<<<1, 256, 0, stream>>>(W_reg, W_tgt, att, wrA, wtA);
  k_attn_proj<<<(Bb * Nn + Bb * Mm) / 8, 256, 0, stream>>>(
      reg_feat, tgt_feat, wrA, wtA, er, etT);
  k_hreg_gemm<<<dim3(Bb * Nn / 128, Hh), 256, 0, stream>>>(reg_feat, W_reg, hrT);
  k_softmax_stats<<<dim3(Mm / 256, Bb), 256, 0, stream>>>(adj, er, etT, lseT,
                                                          adjn);
  k_bit_transpose<<<dim3(Mm / 256, Nn), 256, 0, stream>>>(adjn, adjB);
  k_gat_msgs<<<dim3(Mm / 128, Hh, Nn), 256, 0, stream>>>(hrT, er, etT, lseT,
                                                         adjB, out);
}
